// Glm4MoeAttention_73830487818718
// MI455X (gfx1250) — matrix operand order repaired and checked by reading, not compiled, not run on hardware
//
#include <hip/hip_runtime.h>

// ---------------- problem constants ----------------
constexpr int B_   = 2;
constexpr int S_   = 2048;
constexpr int H_   = 4096;
constexpr int NH_  = 32;
constexpr int NKV_ = 8;
constexpr int HD_  = 128;
constexpr int RD_  = 64;

typedef __bf16 bf16_t;
typedef __attribute__((ext_vector_type(8)))  __bf16 v8bf;
typedef __attribute__((ext_vector_type(16))) __bf16 v16bf;
typedef __attribute__((ext_vector_type(8)))  float  v8f;

union U16 { v16bf v; v8bf h[2]; };

static __device__ __forceinline__ v16bf mk16(v8bf lo, v8bf hi) {
  U16 u; u.h[0] = lo; u.h[1] = hi; return u.v;
}

// ---------------- fp32 -> bf16 conversions ----------------
__global__ __launch_bounds__(256) void cvt_f32_bf16(const float* __restrict__ in,
                                                    bf16_t* __restrict__ out, size_t n) {
  size_t tid = (size_t)blockIdx.x * 256 + threadIdx.x;
  if (tid < n) out[tid] = (bf16_t)in[tid];
}

// out[c*R + r] = in[r*C + c]   (store W transposed so GEMM B-frags are contiguous)
__global__ __launch_bounds__(256) void cvtT_f32_bf16(const float* __restrict__ in,
                                                     bf16_t* __restrict__ out, int R, int C) {
  size_t tid = (size_t)blockIdx.x * 256 + threadIdx.x;
  if (tid >= (size_t)R * C) return;
  int r = (int)(tid / C);
  int c = (int)(tid % C);
  out[(size_t)c * R + r] = (bf16_t)in[tid];
}

// V[b][s][kvh][d] -> Vt[b][kvh][d][s]  (keys contiguous => contiguous PV B-frags)
__global__ __launch_bounds__(256) void transpose_v(const bf16_t* __restrict__ Vb,
                                                   bf16_t* __restrict__ Vt) {
  size_t tid = (size_t)blockIdx.x * 256 + threadIdx.x;   // linear over output
  int s   = (int)(tid % S_);
  int d   = (int)((tid / S_) % HD_);
  int kvh = (int)((tid / ((size_t)S_ * HD_)) % NKV_);
  int b   = (int)(tid / ((size_t)S_ * HD_ * NKV_));
  Vt[tid] = Vb[(((size_t)b * S_ + s) * NKV_ + kvh) * HD_ + d];
}

// ---------------- bf16 WMMA GEMM:  C[M,N] = A[M,K] * BT[N,K]^T ----------------
// One wave computes a 16x64 output tile; register double-buffered K loop.
template <bool STORE_BF16>
__global__ __launch_bounds__(256) void gemm_wmma_nt(const bf16_t* __restrict__ A,
                                                    const bf16_t* __restrict__ BT,
                                                    void* __restrict__ Cv,
                                                    int M, int N, int K) {
  const int wid    = blockIdx.x * (blockDim.x >> 5) + (threadIdx.x >> 5);
  const int lane   = threadIdx.x & 31;
  const int lane16 = lane & 15;
  const int half   = lane >> 4;
  const int nt64   = N >> 6;
  const int tm     = wid / nt64;
  const int n0     = (wid % nt64) << 6;
  const int m0     = tm << 4;
  if (m0 >= M) return;

  const bf16_t* arow = A + (size_t)(m0 + lane16) * K;
  const bf16_t* brow[4];
#pragma unroll
  for (int nn = 0; nn < 4; ++nn)
    brow[nn] = BT + (size_t)(n0 + nn * 16 + lane16) * K + 16 * half;

  v8f acc[4];
#pragma unroll
  for (int i = 0; i < 4; ++i)
#pragma unroll
    for (int e = 0; e < 8; ++e) acc[i][e] = 0.f;

  v16bf afc = mk16(*(const v8bf*)(arow + 16 * half),
                   *(const v8bf*)(arow + 16 * half + 8));
  v16bf bfc[4];
#pragma unroll
  for (int nn = 0; nn < 4; ++nn)
    bfc[nn] = mk16(*(const v8bf*)(brow[nn]), *(const v8bf*)(brow[nn] + 8));

  for (int kk = 0; kk < K; kk += 32) {
    int kn = kk + 32;
    if (kn >= K) kn = kk;  // tail reload (values discarded)

    v16bf afn = mk16(*(const v8bf*)(arow + kn + 16 * half),
                     *(const v8bf*)(arow + kn + 16 * half + 8));
    v16bf bfn[4];
#pragma unroll
    for (int nn = 0; nn < 4; ++nn)
      bfn[nn] = mk16(*(const v8bf*)(brow[nn] + kn), *(const v8bf*)(brow[nn] + kn + 8));

    __builtin_prefetch(arow + kk + 64, 0, 1);

#pragma unroll
    for (int nn = 0; nn < 4; ++nn)
      acc[nn] = __builtin_amdgcn_wmma_f32_16x16x32_bf16(false, afc, false, bfc[nn],
                                                        (short)0, acc[nn], false, false);
    afc = afn;
#pragma unroll
    for (int nn = 0; nn < 4; ++nn) bfc[nn] = bfn[nn];
  }

#pragma unroll
  for (int nn = 0; nn < 4; ++nn) {
    const int col = n0 + nn * 16 + lane16;
#pragma unroll
    for (int r = 0; r < 8; ++r) {
      const size_t idx = (size_t)(m0 + half * 8 + r) * N + col;
      if (STORE_BF16) ((bf16_t*)Cv)[idx] = (bf16_t)acc[nn][r];
      else            ((float*)Cv)[idx]  = acc[nn][r];
    }
  }
}

// ---------------- partial interleaved RoPE (in place, bf16, fp32 math) ----------------
__global__ __launch_bounds__(256) void rope_kernel(bf16_t* __restrict__ X,
                                                   const float* __restrict__ cp,
                                                   const float* __restrict__ sp,
                                                   int nheads, int total) {
  int tid = blockIdx.x * 256 + threadIdx.x;
  if (tid >= total) return;
  int j  = tid & 31;
  int h  = (tid >> 5) % nheads;
  int bs = tid / (32 * nheads);
  size_t base = ((size_t)bs * nheads + h) * HD_;
  float c = cp[(size_t)bs * RD_ + j];
  float s = sp[(size_t)bs * RD_ + j];
  float x0 = (float)X[base + 2 * j];
  float x1 = (float)X[base + 2 * j + 1];
  X[base + 2 * j]     = (bf16_t)(x0 * c - x1 * s);
  X[base + 2 * j + 1] = (bf16_t)(x1 * c + x0 * s);
}

// ---------------- causal flash attention, bf16 WMMA, fp32 softmax ----------------
// One wave owns one (b, head, 16-query tile); 4 independent waves per block.
// Per 32-key chunk both the K tile (32x128) and the Vt tile (128x32, keys
// contiguous) are staged into LDS with GLOBAL_LOAD_ASYNC_TO_LDS_B128, double
// buffered: chunk ch+1 is in flight during chunk ch's WMMAs and softmax.
// All fragment builds are contiguous ds_load_b128 pairs.
__global__ __launch_bounds__(128) void attn_wmma(const bf16_t* __restrict__ Qb,
                                                 const bf16_t* __restrict__ Kb,
                                                 const bf16_t* __restrict__ Vt,
                                                 bf16_t* __restrict__ Ob) {
  __shared__ __align__(16) bf16_t Pl[4][16 * 32];        //  4 KB
  __shared__ __align__(16) bf16_t Kl[4][2][32 * 128];    // 64 KB
  __shared__ __align__(16) bf16_t Vl[4][2][128 * 32];    // 64 KB

  const int w      = threadIdx.x >> 5;
  const int lane   = threadIdx.x & 31;
  const int lane16 = lane & 15;
  const int half   = lane >> 4;

  const int QT  = S_ / 16;
  const int id  = blockIdx.x * 4 + w;
  const int qt  = id % QT;
  const int hh  = (id / QT) % NH_;
  const int b   = id / (QT * NH_);
  const int q0  = qt * 16;
  const int kvh = hh >> 2;                    // NH/NKV = 4

  const size_t krstride = (size_t)NKV_ * HD_; // 1024 elements between keys
  const bf16_t* kbase  = Kb + ((size_t)b * S_ * NKV_ + kvh) * HD_;
  const bf16_t* vtbase = Vt + ((size_t)(b * NKV_ + kvh) * HD_) * S_;

  // async-stage the 32-key K and Vt tiles (8 KB each) for chunk `ch`
  auto stage = [&](int ch, int buf) {
    const int kv0 = ch << 5;
    {  // K tile: row = key (32 rows x 256 B), one row per lane
      const bf16_t* g = kbase + (size_t)(kv0 + lane) * krstride;
      unsigned l0 = (unsigned)(unsigned long long)&Kl[w][buf][lane * 128];
#pragma unroll
      for (int i = 0; i < 16; ++i) {
        unsigned long long ga = (unsigned long long)(g + i * 8);
        asm volatile("global_load_async_to_lds_b128 %0, %1, off"
                     :: "v"(l0 + (unsigned)(i * 16)), "v"(ga) : "memory");
      }
    }
    {  // Vt tile: row = d (128 rows x 64 B), 8 rows per async op
      const int coff  = (lane & 3) * 8;   // element column offset within row
      const int rbase = lane >> 2;        // 0..7
      unsigned l0 = (unsigned)(unsigned long long)&Vl[w][buf][0];
#pragma unroll
      for (int i = 0; i < 16; ++i) {
        const int row = i * 8 + rbase;
        unsigned long long ga = (unsigned long long)(vtbase + (size_t)row * S_ + kv0 + coff);
        asm volatile("global_load_async_to_lds_b128 %0, %1, off"
                     :: "v"(l0 + (unsigned)(row * 64 + coff * 2)), "v"(ga) : "memory");
      }
    }
  };

  // resident Q fragments: 16 rows x 128 dims as 4 K=32 fragments
  const bf16_t* qrow = Qb + (((size_t)b * S_ + q0 + lane16) * NH_ + hh) * HD_;
  v16bf aq[4];
#pragma unroll
  for (int dd = 0; dd < 4; ++dd)
    aq[dd] = mk16(*(const v8bf*)(qrow + dd * 32 + 16 * half),
                  *(const v8bf*)(qrow + dd * 32 + 16 * half + 8));

  float mstate[8], lstate[8];
  v8f o[8];
#pragma unroll
  for (int r = 0; r < 8; ++r) { mstate[r] = -1e30f; lstate[r] = 0.f; }
#pragma unroll
  for (int t = 0; t < 8; ++t)
#pragma unroll
    for (int e = 0; e < 8; ++e) o[t][e] = 0.f;

  const int nkeys  = q0 + 16;
  const int chunks = (nkeys + 31) >> 5;
  const float scale = 0.088388347648318447f;  // 1/sqrt(128)

  stage(0, 0);  // prologue

  for (int ch = 0; ch < chunks; ++ch) {
    const int kv0 = ch << 5;
    const int buf = ch & 1;
    const bool have_next = (ch + 1) < chunks;
    if (have_next) stage(ch + 1, buf ^ 1);   // overlap with this chunk's compute

    // async loads complete in order: <=32 drains this chunk's 32 ops while
    // leaving the next chunk's 32 in flight.
    if (have_next) asm volatile("s_wait_asynccnt 32" ::: "memory");
    else           asm volatile("s_wait_asynccnt 0"  ::: "memory");

    // scores: two 16x16 tiles covering 32 keys; K fragments from LDS (contiguous)
    v8f sc[2];
#pragma unroll
    for (int jj = 0; jj < 2; ++jj) {
#pragma unroll
      for (int e = 0; e < 8; ++e) sc[jj][e] = 0.f;
#pragma unroll
      for (int dd = 0; dd < 4; ++dd) {
        const bf16_t* kb = &Kl[w][buf][(jj * 16 + lane16) * 128 + dd * 32 + 16 * half];
        v16bf bk = mk16(*(const v8bf*)kb, *(const v8bf*)(kb + 8));
        sc[jj] = __builtin_amdgcn_wmma_f32_16x16x32_bf16(false, aq[dd], false, bk,
                                                         (short)0, sc[jj], false, false);
      }
    }

    // scale + causal mask (C layout: row = half*8+r, col = lane16)
#pragma unroll
    for (int jj = 0; jj < 2; ++jj)
#pragma unroll
      for (int r = 0; r < 8; ++r) {
        const int key = kv0 + jj * 16 + lane16;
        const int qg  = q0 + half * 8 + r;
        float v = sc[jj][r] * scale;
        sc[jj][r] = (key <= qg) ? v : -1e30f;
      }

    // online softmax: row reductions across the 16 lanes of each half-group
    float alpha[8];
#pragma unroll
    for (int r = 0; r < 8; ++r) {
      float t = fmaxf(sc[0][r], sc[1][r]);
#pragma unroll
      for (int off = 8; off; off >>= 1) t = fmaxf(t, __shfl_xor(t, off, 16));
      float mn = fmaxf(mstate[r], t);
      alpha[r] = __expf(mstate[r] - mn);
      mstate[r] = mn;
    }
#pragma unroll
    for (int jj = 0; jj < 2; ++jj)
#pragma unroll
      for (int r = 0; r < 8; ++r) sc[jj][r] = __expf(sc[jj][r] - mstate[r]);
#pragma unroll
    for (int r = 0; r < 8; ++r) {
      float s = sc[0][r] + sc[1][r];
#pragma unroll
      for (int off = 8; off; off >>= 1) s += __shfl_xor(s, off, 16);
      lstate[r] = lstate[r] * alpha[r] + s;
    }
#pragma unroll
    for (int t = 0; t < 8; ++t)
#pragma unroll
      for (int r = 0; r < 8; ++r) o[t][r] *= alpha[r];

    // P (C layout) -> LDS -> A-fragment layout (LDS ops are in-order per wave)
#pragma unroll
    for (int jj = 0; jj < 2; ++jj)
#pragma unroll
      for (int r = 0; r < 8; ++r)
        Pl[w][(half * 8 + r) * 32 + jj * 16 + lane16] = (bf16_t)sc[jj][r];
    asm volatile("s_wait_dscnt 0" ::: "memory");

    v16bf pf = mk16(*(const v8bf*)&Pl[w][lane16 * 32 + 16 * half],
                    *(const v8bf*)&Pl[w][lane16 * 32 + 16 * half + 8]);

    // O += P(16x32) @ V(32x16): Vt rows are d, keys contiguous -> ds_b128 pairs
#pragma unroll
    for (int t = 0; t < 8; ++t) {
      const bf16_t* vb = &Vl[w][buf][(t * 16 + lane16) * 32 + 16 * half];
      v16bf vf = mk16(*(const v8bf*)vb, *(const v8bf*)(vb + 8));
      o[t] = __builtin_amdgcn_wmma_f32_16x16x32_bf16(false, pf, false, vf,
                                                     (short)0, o[t], false, false);
    }
    asm volatile("s_wait_dscnt 0" ::: "memory");  // drain reads before buffer reuse
  }

  float inv[8];
#pragma unroll
  for (int r = 0; r < 8; ++r) inv[r] = 1.0f / lstate[r];
#pragma unroll
  for (int t = 0; t < 8; ++t)
#pragma unroll
    for (int r = 0; r < 8; ++r)
      Ob[(((size_t)b * S_ + q0 + half * 8 + r) * NH_ + hh) * HD_ + t * 16 + lane16] =
          (bf16_t)(o[t][r] * inv[r]);
}

// ---------------- host orchestration ----------------
extern "C" void kernel_launch(void* const* d_in, const int* in_sizes, int n_in,
                              void* d_out, int out_size, void* d_ws, size_t ws_size,
                              hipStream_t stream) {
  (void)in_sizes; (void)n_in; (void)out_size; (void)ws_size;

  const float* hs   = (const float*)d_in[0];
  const float* wq   = (const float*)d_in[1];
  const float* wk   = (const float*)d_in[2];
  const float* wv   = (const float*)d_in[3];
  const float* wo   = (const float*)d_in[4];
  const float* cosp = (const float*)d_in[5];
  const float* sinp = (const float*)d_in[6];

  const size_t M  = (size_t)B_ * S_;      // 4096
  const size_t QN = (size_t)NH_ * HD_;    // 4096
  const size_t KN = (size_t)NKV_ * HD_;   // 1024

  char* ws = (char*)d_ws;
  size_t off = 0;
  bf16_t* Xb  = (bf16_t*)(ws + off); off += M * H_ * 2;            // 32 MB
  bf16_t* WqT = (bf16_t*)(ws + off); off += (size_t)H_ * QN * 2;   // 32 MB
  bf16_t* WkT = (bf16_t*)(ws + off); off += (size_t)H_ * KN * 2;   //  8 MB
  bf16_t* WvT = (bf16_t*)(ws + off); off += (size_t)H_ * KN * 2;   //  8 MB
  bf16_t* WoT = (bf16_t*)(ws + off); off += QN * (size_t)H_ * 2;   // 32 MB
  bf16_t* Qb  = (bf16_t*)(ws + off); off += M * QN * 2;            // 32 MB
  bf16_t* Kb  = (bf16_t*)(ws + off); off += M * KN * 2;            //  8 MB
  bf16_t* Vb  = (bf16_t*)(ws + off); off += M * KN * 2;            //  8 MB
  bf16_t* Vtb = (bf16_t*)(ws + off); off += M * KN * 2;            //  8 MB
  bf16_t* Ob  = (bf16_t*)(ws + off); off += M * QN * 2;            // 32 MB

  // 1) precision conversion (weights transposed for contiguous WMMA B-frags)
  cvt_f32_bf16<<<(unsigned)((M * H_) / 256), 256, 0, stream>>>(hs, Xb, M * H_);
  cvtT_f32_bf16<<<(unsigned)(((size_t)H_ * QN) / 256), 256, 0, stream>>>(wq, WqT, H_, (int)QN);
  cvtT_f32_bf16<<<(unsigned)(((size_t)H_ * KN) / 256), 256, 0, stream>>>(wk, WkT, H_, (int)KN);
  cvtT_f32_bf16<<<(unsigned)(((size_t)H_ * KN) / 256), 256, 0, stream>>>(wv, WvT, H_, (int)KN);
  cvtT_f32_bf16<<<(unsigned)(QN * (size_t)H_ / 256), 256, 0, stream>>>(wo, WoT, (int)QN, H_);

  // 2) QKV projections (bf16 WMMA, bf16 out)
  {
    unsigned wavesQ = (unsigned)((M / 16) * (QN / 64));   // 16384
    unsigned wavesK = (unsigned)((M / 16) * (KN / 64));   // 4096
    gemm_wmma_nt<true><<<wavesQ / 8, 256, 0, stream>>>(Xb, WqT, Qb, (int)M, (int)QN, H_);
    gemm_wmma_nt<true><<<wavesK / 8, 256, 0, stream>>>(Xb, WkT, Kb, (int)M, (int)KN, H_);
    gemm_wmma_nt<true><<<wavesK / 8, 256, 0, stream>>>(Xb, WvT, Vb, (int)M, (int)KN, H_);
  }

  // 3) partial interleaved RoPE on Q and K; V transposed to [b][kvh][d][s]
  {
    int totQ = (int)(M * NH_ * (RD_ / 2));
    int totK = (int)(M * NKV_ * (RD_ / 2));
    rope_kernel<<<totQ / 256, 256, 0, stream>>>(Qb, cosp, sinp, NH_, totQ);
    rope_kernel<<<totK / 256, 256, 0, stream>>>(Kb, cosp, sinp, NKV_, totK);
    transpose_v<<<(unsigned)((M * KN) / 256), 256, 0, stream>>>(Vb, Vtb);
  }

  // 4) causal flash attention (async double-buffered K/Vt staging)
  {
    unsigned tiles = (unsigned)(B_ * NH_ * (S_ / 16));    // 8192 waves
    attn_wmma<<<tiles / 4, 128, 0, stream>>>(Qb, Kb, Vtb, Ob);
  }

  // 5) output projection, fp32 result
  {
    unsigned wavesO = (unsigned)((M / 16) * ((size_t)H_ / 64));  // 16384
    gemm_wmma_nt<false><<<wavesO / 8, 256, 0, stream>>>(Ob, WoT, (float*)d_out,
                                                        (int)M, H_, (int)QN);
  }
}
